// MixedScoreMultiHeadAttention_56659208569415
// MI455X (gfx1250) — matrix operand order repaired and checked by reading, not compiled, not run on hardware
//
// MixedScoreMultiHeadAttention for MI455X (gfx1250), wave32 + WMMA bf16.
//
// b=4, r=c=512, H=8, D=64, E=512, MS=16.
// Compute-bound (~13 GFLOP vs ~45MB traffic -> HBM time ~2us @23.3TB/s), so
// all GEMM-shaped math uses v_wmma_f32_16x16x32_bf16; the tiny K=16 mixer MLP
// runs on VALU with weights staged in LDS (cost-column of Wmix1 pre-folded).
// The fused attention block keeps 8 heads x 16 rows x 512 cols of f32 logits
// in 256KB LDS (CDNA5's 320KB/WGP makes this single-pass softmax possible).
// GEMMs use 32x64 register blocking per wave: 2 A-frags x 4 B-frags -> 8 WMMA
// per K-step at 24B/lane/WMMA of fragment traffic (vs 64B unblocked).

#include <hip/hip_runtime.h>
#include <hip/hip_bf16.h>
#include <cstdint>

typedef __attribute__((ext_vector_type(16))) __bf16          v16bf;
typedef __attribute__((ext_vector_type(8)))  float           v8f;
typedef __attribute__((ext_vector_type(8)))  unsigned short  v8u;
typedef __attribute__((ext_vector_type(16))) unsigned short  v16u;
typedef __attribute__((ext_vector_type(4)))  unsigned short  v4u;

__device__ __forceinline__ unsigned short f2bf_bits(float f) {
  unsigned u = __builtin_bit_cast(unsigned, f);
  u += 0x7FFFu + ((u >> 16) & 1u);          // round-to-nearest-even
  return (unsigned short)(u >> 16);
}
__device__ __forceinline__ __bf16 f2bf(float f) {
  unsigned short s = f2bf_bits(f);
  return __builtin_bit_cast(__bf16, s);
}

// ---- WMMA bf16 fragment loads (wave32; contiguous k order, not the ISA table order, same for A and B) ----
// A 16x32: lane m=lane&15, sel=lane>>4. a[0..15]=K=kk+16*sel..+15.
// p must already include (kk + 16*sel) elements of offset; halves are 16B loads.
__device__ __forceinline__ v16bf lda16(const __bf16* p) {
  v8u lo = *(const v8u*)(p);
  v8u hi = *(const v8u*)(p + 8);
  v16u t = __builtin_shufflevector(lo, hi, 0,1,2,3,4,5,6,7,8,9,10,11,12,13,14,15);
  return __builtin_bit_cast(v16bf, t);
}
// B 32x16: lane n=lane&15, sel=lane>>4. b[0..15]=K=kk+16*sel..+15 (32B load).
__device__ __forceinline__ v16bf ldb16(const __bf16* p) {
  return __builtin_bit_cast(v16bf, *(const v16u*)(p));
}

// ------------------------------ convert kernel ------------------------------
__global__ void f32_to_bf16_kernel(const float* __restrict__ in,
                                   __bf16* __restrict__ out, int n4) {
  int i = blockIdx.x * blockDim.x + threadIdx.x;
  if (i >= n4) return;
  float4 v = ((const float4*)in)[i];
  v4u o;
  o[0] = f2bf_bits(v.x); o[1] = f2bf_bits(v.y);
  o[2] = f2bf_bits(v.z); o[3] = f2bf_bits(v.w);
  ((v4u*)out)[i] = o;
}

// ------------------------------- GEMM kernel --------------------------------
// C[n,o] = sum_e A[n,e]*W[o,e] ; A,W bf16 row-major [rows x 512].
// One wave computes a 32x64 block (2 M-tiles x 4 N-tiles): each A fragment is
// reused by 4 WMMAs, each B fragment by 2 -> 8 WMMA per 12 b128-loads per lane.
// MODE 0: store bf16 [n,o]. MODE 1: store bf16 transposed-V layout [b,h,d,c].
// MODE 2: store f32 [n,o] (final output).
template <int MODE>
__global__ void gemm_wmma_kernel(const __bf16* __restrict__ A,
                                 const __bf16* __restrict__ W,
                                 void* __restrict__ C, int n_rows) {
  const int lane = threadIdx.x & 31;
  const int wave = threadIdx.x >> 5;
  const int sel  = lane >> 4;
  const int mn   = lane & 15;
  const int job  = blockIdx.x * 8 + wave;
  const int tm   = job >> 3;           // row-group of 32 (64 groups for 2048)
  const int tn   = job & 7;            // col-group of 64 (8 groups for 512)
  if (tm * 32 >= n_rows) return;       // wave-uniform

  const __bf16* arow0 = A + (size_t)(tm * 32 + mn) * 512;
  const __bf16* arow1 = arow0 + (size_t)16 * 512;
  const __bf16* wbase = W + (size_t)(tn * 64 + mn) * 512;

  v8f acc[2][4] = {};
#pragma unroll 4
  for (int kk = 0; kk < 512; kk += 32) {
    __builtin_prefetch(arow0 + kk + 128, 0, 1);   // global_prefetch_b8
    v16bf a0 = lda16(arow0 + kk + 16 * sel);
    v16bf a1 = lda16(arow1 + kk + 16 * sel);
#pragma unroll
    for (int nt = 0; nt < 4; ++nt) {
      v16bf b = ldb16(wbase + (size_t)nt * 16 * 512 + kk + 16 * sel);
      acc[0][nt] = __builtin_amdgcn_wmma_f32_16x16x32_bf16(
          false, a0, false, b, (short)0, acc[0][nt], false, false);
      acc[1][nt] = __builtin_amdgcn_wmma_f32_16x16x32_bf16(
          false, a1, false, b, (short)0, acc[1][nt], false, false);
    }
  }

#pragma unroll
  for (int mi = 0; mi < 2; ++mi)
#pragma unroll
    for (int nt = 0; nt < 4; ++nt) {
      const int col = tn * 64 + nt * 16 + mn;
#pragma unroll
      for (int i = 0; i < 8; ++i) {
        const int gr = tm * 32 + mi * 16 + i + 8 * sel;  // C: M = i + 8*sel
        const float v = acc[mi][nt][i];
        if (MODE == 2) {
          ((float*)C)[(size_t)gr * 512 + col] = v;
        } else if (MODE == 0) {
          ((__bf16*)C)[(size_t)gr * 512 + col] = f2bf(v);
        } else {                                     // V -> Vt[b][h][d][c]
          const int bb = gr >> 9, cc = gr & 511;
          const int hh = col >> 6, dd = col & 63;
          ((__bf16*)C)[((size_t)(bb * 8 + hh) * 64 + dd) * 512 + cc] = f2bf(v);
        }
      }
    }
}

// ---------------------------- fused attention -------------------------------
__device__ __forceinline__ bool all4b(unsigned v) {
  return (v & 0xFFu) && (v & 0xFF00u) && (v & 0xFF0000u) && (v & 0xFF000000u);
}

// One block per (batch, 16-row tile). 8 waves = 8 heads. Dynamic LDS:
//   sL  [8*16*512] f32 logits/probs (256KB)
//   sW1 [128*16], sW2t [128*8] (transposed), sC1 [128]
__global__ void attn_fused_kernel(const __bf16* __restrict__ Qb,
                                  const __bf16* __restrict__ Kb,
                                  const __bf16* __restrict__ Vt,
                                  const float* __restrict__ cost,
                                  const unsigned char* __restrict__ maskb,
                                  const float* __restrict__ W1,
                                  const float* __restrict__ W2,
                                  __bf16* __restrict__ attn_out) {
  extern __shared__ float smem[];
  float* sL  = smem;                 // 65536
  float* sW1 = sL + 8 * 16 * 512;    // 2048
  float* sW2 = sW1 + 2048;           // 1024, [j][h]
  float* sC1 = sW2 + 1024;           // 128
  __shared__ unsigned char sAllM[128];

  const int tid  = threadIdx.x;
  const int lane = tid & 31;
  const int h    = tid >> 5;         // wave == head
  const int sel  = lane >> 4;
  const int mn   = lane & 15;
  const int bb   = blockIdx.x >> 5;  // batch
  const int rt   = blockIdx.x & 31;  // row tile
  const int r0   = rt * 16;

  // ---- phase 0: stage mixer weights (fold cost columns of W1 into sC1) ----
  for (int i = tid; i < 2048; i += 256) sW1[i] = W1[i];
  for (int i = tid; i < 1024; i += 256) {
    const int hh = i >> 7, j = i & 127;
    sW2[j * 8 + hh] = W2[i];
  }
  for (int j = tid; j < 128; j += 256) {
    float s = 0.f;
#pragma unroll
    for (int hh = 0; hh < 8; ++hh) s += W1[j * 16 + 2 * hh + 1];
    sC1[j] = s;
  }

  // ---- phase 1: dot = norm * q.k^T for this wave's head, into LDS ----
  const __bf16* qrow = Qb + ((size_t)(bb * 512 + r0 + mn)) * 512 + h * 64;
  for (int ct = 0; ct < 32; ++ct) {
    const __bf16* krow = Kb + ((size_t)(bb * 512 + ct * 16 + mn)) * 512 + h * 64;
    v8f acc = {};
#pragma unroll
    for (int kk = 0; kk < 64; kk += 32) {
      v16bf a = lda16(qrow + kk + 16 * sel);
      v16bf b = ldb16(krow + kk + 16 * sel);
      acc = __builtin_amdgcn_wmma_f32_16x16x32_bf16(
          false, a, false, b, (short)0, acc, false, false);
    }
#pragma unroll
    for (int i = 0; i < 8; ++i)
      sL[(size_t)(h * 16 + i + 8 * sel) * 512 + ct * 16 + mn] = 0.125f * acc[i];
  }

  // ---- per-row "fully masked" flags for this head ----
  const unsigned char* mrow =
      maskb + ((size_t)(bb * 8 + h) * 512 + r0) * 512;
  for (int m = 0; m < 16; ++m) {
    const uint4 mv = *(const uint4*)(mrow + (size_t)m * 512 + lane * 16);
    bool a = all4b(mv.x) && all4b(mv.y) && all4b(mv.z) && all4b(mv.w);
    a = __all(a);
    if (lane == 0) sAllM[h * 16 + m] = a ? 1 : 0;
  }
  __syncthreads();

  // ---- phase 2: cross-head mixer MLP + mask, in place in LDS ----
  const float* costrow = cost + ((size_t)bb * 512 + r0) * 512;
  for (int e = tid; e < 16 * 512; e += 256) {
    const int m = e >> 9, c = e & 511;
    float dv[8];
#pragma unroll
    for (int hh = 0; hh < 8; ++hh) dv[hh] = sL[(size_t)(hh * 16 + m) * 512 + c];
    const float cs = costrow[(size_t)m * 512 + c];
    float ms[8] = {0, 0, 0, 0, 0, 0, 0, 0};
    for (int j = 0; j < 128; ++j) {
      const float* w1 = sW1 + j * 16;
      float t = cs * sC1[j];
#pragma unroll
      for (int hh = 0; hh < 8; ++hh) t += w1[2 * hh] * dv[hh];
      t = fmaxf(t, 0.f);
      const float* w2 = sW2 + j * 8;
#pragma unroll
      for (int hh = 0; hh < 8; ++hh) ms[hh] += w2[hh] * t;
    }
#pragma unroll
    for (int hh = 0; hh < 8; ++hh) {
      const bool mk =
          maskb[((size_t)(bb * 8 + hh) * 512 + r0 + m) * 512 + c] != 0;
      const bool am = sAllM[hh * 16 + m] != 0;
      sL[(size_t)(hh * 16 + m) * 512 + c] =
          (mk && !am) ? -__builtin_inff() : ms[hh];
    }
  }
  __syncthreads();

  // ---- phase 3: softmax over c (wave h handles head h) ----
  for (int m = 0; m < 16; ++m) {
    float* row = sL + (size_t)(h * 16 + m) * 512;
    float mx = -__builtin_inff();
    for (int c = lane; c < 512; c += 32) mx = fmaxf(mx, row[c]);
#pragma unroll
    for (int off = 16; off > 0; off >>= 1)
      mx = fmaxf(mx, __shfl_xor(mx, off, 32));
    float sum = 0.f;
    for (int c = lane; c < 512; c += 32) {
      const float ev = __expf(row[c] - mx);
      row[c] = ev;
      sum += ev;
    }
#pragma unroll
    for (int off = 16; off > 0; off >>= 1) sum += __shfl_xor(sum, off, 32);
    const float inv = 1.f / sum;
    for (int c = lane; c < 512; c += 32) row[c] *= inv;
  }
  __syncthreads();

  // ---- phase 4: out[h,16r,64d] = P @ V (WMMA over c, Vt is [b,h,d,c]) ----
  v8f oacc[4] = {v8f{}, v8f{}, v8f{}, v8f{}};
  const float* prow = sL + (size_t)(h * 16 + mn) * 512;
  for (int kk = 0; kk < 512; kk += 32) {
    v16bf a;
    const float* p0 = prow + kk + 16 * sel;
#pragma unroll
    for (int i = 0; i < 8; ++i) a[i] = f2bf(p0[i]);
#pragma unroll
    for (int i = 0; i < 8; ++i) a[8 + i] = f2bf(p0[8 + i]);
#pragma unroll
    for (int nt = 0; nt < 4; ++nt) {
      const __bf16* vb =
          Vt + ((size_t)((bb * 8 + h) * 64 + nt * 16 + mn)) * 512 + kk + 16 * sel;
      oacc[nt] = __builtin_amdgcn_wmma_f32_16x16x32_bf16(
          false, a, false, ldb16(vb), (short)0, oacc[nt], false, false);
    }
  }
#pragma unroll
  for (int nt = 0; nt < 4; ++nt)
#pragma unroll
    for (int i = 0; i < 8; ++i) {
      const int rr = r0 + i + 8 * sel;
      attn_out[((size_t)bb * 512 + rr) * 512 + h * 64 + nt * 16 + mn] =
          f2bf(oacc[nt][i]);
    }
}

// ------------------------------ host launcher -------------------------------
extern "C" void kernel_launch(void* const* d_in, const int* in_sizes, int n_in,
                              void* d_out, int out_size, void* d_ws,
                              size_t ws_size, hipStream_t stream) {
  const float* row_emb = (const float*)d_in[0];         // [4,512,512]
  const float* col_emb = (const float*)d_in[1];         // [4,512,512]
  const float* cost    = (const float*)d_in[2];         // [4,512,512]
  const unsigned char* mask = (const unsigned char*)d_in[3]; // bool [4,8,512,512]
  const float* Wq   = (const float*)d_in[4];            // [512,512]
  const float* Wk   = (const float*)d_in[5];
  const float* Wv   = (const float*)d_in[6];
  const float* Wm1  = (const float*)d_in[7];            // [128,16]
  const float* Wm2  = (const float*)d_in[8];            // [8,128]
  const float* Wout = (const float*)d_in[9];
  float* out = (float*)d_out;                           // [2048,512] f32

  char* ws = (char*)d_ws;
  const size_t SZ_ACT = (size_t)2048 * 512 * 2;         // 2MB bf16
  const size_t SZ_W   = (size_t)512 * 512 * 2;          // 512KB bf16
  __bf16* rowb  = (__bf16*)(ws);
  __bf16* colb  = (__bf16*)(ws + SZ_ACT);
  __bf16* Wqb   = (__bf16*)(ws + 2 * SZ_ACT);
  __bf16* Wkb   = (__bf16*)(ws + 2 * SZ_ACT + SZ_W);
  __bf16* Wvb   = (__bf16*)(ws + 2 * SZ_ACT + 2 * SZ_W);
  __bf16* Woutb = (__bf16*)(ws + 2 * SZ_ACT + 3 * SZ_W);
  __bf16* Qb    = (__bf16*)(ws + 2 * SZ_ACT + 4 * SZ_W);
  __bf16* Kb    = (__bf16*)(ws + 3 * SZ_ACT + 4 * SZ_W);
  __bf16* Vt    = (__bf16*)(ws + 4 * SZ_ACT + 4 * SZ_W);
  __bf16* AOb   = (__bf16*)(ws + 5 * SZ_ACT + 4 * SZ_W);

  // f32 -> bf16 conversions (all element counts divisible by 1024)
  f32_to_bf16_kernel<<<1024, 256, 0, stream>>>(row_emb, rowb, 262144);
  f32_to_bf16_kernel<<<1024, 256, 0, stream>>>(col_emb, colb, 262144);
  f32_to_bf16_kernel<<<256, 256, 0, stream>>>(Wq, Wqb, 65536);
  f32_to_bf16_kernel<<<256, 256, 0, stream>>>(Wk, Wkb, 65536);
  f32_to_bf16_kernel<<<256, 256, 0, stream>>>(Wv, Wvb, 65536);
  f32_to_bf16_kernel<<<256, 256, 0, stream>>>(Wout, Woutb, 65536);

  // projections: 2048x512 @ 512x512^T ; (2048/32)*(512/64)=512 wave-jobs,
  // 8 waves/block -> 64 blocks
  gemm_wmma_kernel<0><<<64, 256, 0, stream>>>(rowb, Wqb, Qb, 2048);
  gemm_wmma_kernel<0><<<64, 256, 0, stream>>>(colb, Wkb, Kb, 2048);
  gemm_wmma_kernel<1><<<64, 256, 0, stream>>>(colb, Wvb, Vt, 2048);

  // fused attention: 128 blocks (4 batches x 32 row-tiles), 8 waves each,
  // 274944B dynamic LDS (needs CDNA5's 320KB WGP LDS)
  const int smem_bytes = (8 * 16 * 512 + 2048 + 1024 + 128) * 4;
  (void)hipFuncSetAttribute((const void*)attn_fused_kernel,
                            hipFuncAttributeMaxDynamicSharedMemorySize,
                            smem_bytes);
  attn_fused_kernel<<<128, 256, smem_bytes, stream>>>(Qb, Kb, Vt, cost, mask,
                                                      Wm1, Wm2, AOb);

  // final projection to f32 output
  gemm_wmma_kernel<2><<<64, 256, 0, stream>>>(AOb, Woutb, out, 2048);
}
